// HeadAttentionMulti_75419625717761
// MI455X (gfx1250) — compile-verified, matrix operand order checked by reading, not run on hardware
//
#include <hip/hip_runtime.h>
#include <hip/hip_bf16.h>

#define BB 8
#define SS 1024
#define HH 16
#define DEPTH 64
#define DMODEL 1024

typedef __bf16 bf16;
typedef __attribute__((ext_vector_type(16))) __bf16 v16bf;
typedef __attribute__((ext_vector_type(8)))  float  v8f;

// ---------------------------------------------------------------------------
// Async global->LDS staging (CDNA5 GLOBAL_LOAD_ASYNC_TO_LDS_B128, ASYNCcnt),
// with a synchronous uint4 fallback if the toolchain lacks the builtins.
// The builtin takes pointers to 16B int vectors in AS1 (global) / AS3 (LDS).
// ---------------------------------------------------------------------------
#if defined(__has_builtin)
#if __has_builtin(__builtin_amdgcn_global_load_async_to_lds_b128) && \
    __has_builtin(__builtin_amdgcn_s_wait_asynccnt)
#define USE_ASYNC_LDS 1
#endif
#endif

typedef __attribute__((__vector_size__(4 * sizeof(int)))) int i32x4;
typedef __attribute__((address_space(1))) i32x4 gi32x4;
typedef __attribute__((address_space(3))) i32x4 li32x4;

__device__ __forceinline__ void stage_b128(const void* g, void* l) {
#ifdef USE_ASYNC_LDS
  __builtin_amdgcn_global_load_async_to_lds_b128((gi32x4*)g, (li32x4*)l, 0, 0);
#else
  *(uint4*)l = *(const uint4*)g;
#endif
}

__device__ __forceinline__ void stage_wait() {
#ifdef USE_ASYNC_LDS
  __builtin_amdgcn_s_wait_asynccnt(0);
#endif
  __syncthreads();
}

// ---------------------------------------------------------------------------
// Fragment-order layouts (CDNA5 wmma 16x16x32 bf16, wave32; ISA 7.12.2):
// A-fragment of lane L: row r = L&15, kb = (L>>4)*8, elems j=0..7 -> K=kb+j,
// j=8..15 -> K=kb+8+j.  LDS tiles stored as [subtile][lane][16] so every
// fragment is one contiguous 32B read (2x ds_load_b128).
// A 16x32 row-major row r splits into 4 16B chunks c=0..3:
//   chunk c -> lane = r + ((c&1)<<4), j-base = (c>>1)<<3.
// ---------------------------------------------------------------------------

// fp32 -> bf16 cast (grid-stride), for the activation matrix x
__global__ void cast_f32_to_bf16(const float* __restrict__ in,
                                 bf16* __restrict__ out, int n) {
  int i = blockIdx.x * blockDim.x + threadIdx.x;
  const int stride = gridDim.x * blockDim.x;
  for (; i < n; i += stride) out[i] = (bf16)in[i];
}

// Pack a 1024x1024 f32 weight matrix into fragment-major bf16:
// idx = ((k0/32)*64 + (n0/16))*512 + lane*16 + j.
__global__ __launch_bounds__(256) void pack_w_bf16(const float* __restrict__ W,
                                                   bf16* __restrict__ out) {
  const int t = blockIdx.x * blockDim.x + threadIdx.x;  // 0..65535
  const int lane  = t & 31;
  const int tileN = (t >> 5) & 63;
  const int tileK = t >> 11;
  const int nn = lane & 15;
  const int kb = (lane >> 4) << 3;
  const int k0 = tileK << 5, n0 = tileN << 4;
  v16bf f;
#pragma unroll
  for (int j = 0; j < 16; ++j) {
    const int kk = (j < 8) ? (kb + j) : (kb + j + 8);
    f[j] = (bf16)W[(size_t)(k0 + kk) * DMODEL + n0 + nn];
  }
  *(v16bf*)&out[(size_t)t << 4] = f;
}

// ---------------------------------------------------------------------------
// GEMM: (8192 x 1024) @ (1024 x 1024) + bias.  Block tile 128x128, 8 waves,
// wave = 32 rows x 64 cols -> 8 wmma per k-step from 12 LDS b128 frag loads.
// headSplit: (B,H,S,DEPTH) output layout.
// bfMode: 0 = no bf16 mirror, 1 = mirror at same index, 2 = V-packed mirror.
// ---------------------------------------------------------------------------
__global__ __launch_bounds__(256) void gemm_bf16(
    const bf16* __restrict__ A, const bf16* __restrict__ Wpk,
    const float* __restrict__ bias, float* __restrict__ outF,
    bf16* __restrict__ outBf, int headSplit, int bfMode) {
  constexpr int K = DMODEL, N = DMODEL;
  __shared__ __align__(32) bf16 lA[8 * 512];  // 128x32, fragment order
  __shared__ __align__(32) bf16 lB[8 * 512];  // 32x128, fragment order
  const int tid  = threadIdx.x;
  const int wave = tid >> 5, lane = tid & 31;
  const int m0 = blockIdx.x * 128;
  const int n0 = blockIdx.y * 128;
  const int wm = wave >> 1;  // 0..3: rows wm*32
  const int wn = wave & 1;   // 0..1: cols wn*64

  v8f acc[2][4] = {};
  for (int k0 = 0; k0 < K; k0 += 32) {
    // stage A with fragment swizzle: 512 chunks of 16B, 2 per thread
#pragma unroll
    for (int i = 0; i < 2; ++i) {
      const int c4 = tid + (i << 8);
      const int r = c4 >> 2, c = c4 & 3;
      const int dst = ((r >> 4) << 9) + (((r & 15) + ((c & 1) << 4)) << 4) +
                      ((c >> 1) << 3);
      stage_b128(A + (size_t)(m0 + r) * K + k0 + (c << 3), &lA[dst]);
    }
    // stage B: 8KB straight copy (already fragment-major)
    const bf16* wsrc = Wpk + ((((size_t)k0 >> 5) << 6) + (n0 >> 4)) * 512;
#pragma unroll
    for (int i = 0; i < 2; ++i) {
      const int off = (tid + (i << 8)) << 3;
      stage_b128(wsrc + off, &lB[off]);
    }
    if (k0 + 32 < K) {
      __builtin_prefetch(A + (size_t)(m0 + (tid >> 1)) * K + k0 + 32, 0, 1);
      __builtin_prefetch(wsrc + (64 * 512) + (tid << 3), 0, 1);
      __builtin_prefetch(wsrc + (64 * 512) + 2048 + (tid << 3), 0, 1);
    }
    stage_wait();
    v16bf afr[2];
#pragma unroll
    for (int mt = 0; mt < 2; ++mt)
      afr[mt] = *(const v16bf*)&lA[(((wm << 1) + mt) << 9) + (lane << 4)];
#pragma unroll
    for (int nt = 0; nt < 4; ++nt) {
      const v16bf bfr =
          *(const v16bf*)&lB[(((wn << 2) + nt) << 9) + (lane << 4)];
#pragma unroll
      for (int mt = 0; mt < 2; ++mt)
        acc[mt][nt] = __builtin_amdgcn_wmma_f32_16x16x32_bf16(
            false, afr[mt], false, bfr, (short)0, acc[mt][nt], false, false);
    }
    __syncthreads();
  }

#pragma unroll
  for (int mt = 0; mt < 2; ++mt) {
#pragma unroll
    for (int nt = 0; nt < 4; ++nt) {
      const int n = n0 + (wn << 6) + (nt << 4) + (lane & 15);
      const float bv = bias[n];
#pragma unroll
      for (int r = 0; r < 8; ++r) {
        const int m = m0 + (((wm << 1) + mt) << 4) + r + ((lane >> 4) << 3);
        const float val = acc[mt][nt][r] + bv;
        int bb_ = 0, s_ = 0, h_ = 0, dep_ = 0;
        size_t idx;
        if (headSplit) {
          bb_ = m >> 10; s_ = m & 1023; h_ = n >> 6; dep_ = n & 63;
          idx = ((size_t)(bb_ * HH + h_) * SS + s_) * DEPTH + dep_;
        } else {
          idx = (size_t)m * N + n;
        }
        outF[idx] = val;
        if (outBf) {
          size_t bidx = idx;
          if (bfMode == 2) {  // V packed fragment-major per (b,h)
            const int tk = s_ >> 5, ks = s_ & 31;
            const int hi = (ks >> 3) & 1;
            const int jj = (ks & 7) | ((ks >> 4) << 3);
            bidx = ((size_t)(bb_ * HH + h_) << 16) + (tk << 11) +
                   ((dep_ >> 4) << 9) + (((hi << 4) + (dep_ & 15)) << 4) + jj;
          }
          outBf[bidx] = (bf16)val;
        }
      }
    }
  }
}

// ---------------------------------------------------------------------------
// Attention per (b, h, 64-row q block).  LDS-resident 64x1024 f32 logits.
// Q/K tiles staged async in fragment order (same swizzle serves Q as A-frag
// and K as B=K^T-frag).  V arrives pre-packed fragment-major from QKV GEMM.
// ---------------------------------------------------------------------------
__global__ __launch_bounds__(256) void attn_kernel(
    const bf16* __restrict__ qbf, const bf16* __restrict__ kbf,
    const bf16* __restrict__ vpk, const float* __restrict__ mask,
    float* __restrict__ attnOut, float* __restrict__ qkOut,
    bf16* __restrict__ zbf) {
  extern __shared__ char smem[];
  float* logits = (float*)smem;                  // 64*1024*4 = 262144 B
  bf16* qtile = (bf16*)(smem + 262144);          // [rt][ks][lane][16] 8192 B
  bf16* ktile = (bf16*)(smem + 262144 + 8192);   // 8192 B
  bf16* vtile = (bf16*)(smem + 262144 + 16384);  // 4096 B (packed)
  float* red  = (float*)(smem + 262144 + 20480); // 2048 B

  const int tid  = threadIdx.x;
  const int wave = tid >> 5, lane = tid & 31;
  const int sq0 = blockIdx.x * 64;
  const int h = blockIdx.y, b = blockIdx.z;
  const size_t bh = (size_t)(b * HH + h);
  const bf16* Q  = qbf + (bh * SS + sq0) * DEPTH;
  const bf16* Kp = kbf + bh * SS * DEPTH;
  const bf16* Vp = vpk + (bh << 16);

  // stage Q tile 64x64 with fragment swizzle (512 chunks, 2 per thread)
#pragma unroll
  for (int i = 0; i < 2; ++i) {
    const int c4 = tid + (i << 8);
    const int r = c4 >> 3, c = c4 & 7;
    const int ks = c >> 2, cc = c & 3;
    const int dst = ((r >> 4) << 10) + (ks << 9) +
                    (((r & 15) + ((cc & 1) << 4)) << 4) + ((cc >> 1) << 3);
    stage_b128(Q + (size_t)r * DEPTH + (c << 3), &qtile[dst]);
  }

  const int rt    = wave >> 1;       // wave's 16-row tile (0..3)
  const int ctb   = (wave & 1) * 2;  // wave's first 16-col tile
  const int rhalf = (lane >> 4) << 3;
  const int lcol  = lane & 15;

  // ---- logits = scale * Q K^T, masked, into LDS; qk_values to HBM ----
  for (int kb0 = 0; kb0 < SS; kb0 += 64) {
#pragma unroll
    for (int i = 0; i < 2; ++i) {
      const int c4 = tid + (i << 8);
      const int r = c4 >> 3, c = c4 & 7;
      const int ks = c >> 2, cc = c & 3;
      const int dst = ((r >> 4) << 10) + (ks << 9) +
                      (((r & 15) + ((cc & 1) << 4)) << 4) + ((cc >> 1) << 3);
      stage_b128(Kp + (size_t)(kb0 + r) * DEPTH + (c << 3), &ktile[dst]);
    }
    if (kb0 + 64 < SS)
      __builtin_prefetch(Kp + (size_t)(kb0 + 64 + (tid >> 2)) * DEPTH, 0, 1);
    stage_wait();
#pragma unroll
    for (int c = 0; c < 2; ++c) {
      const int ct = ctb + c;
      v8f acc = {};
#pragma unroll
      for (int ks = 0; ks < 2; ++ks) {
        const v16bf a =
            *(const v16bf*)&qtile[(rt << 10) + (ks << 9) + (lane << 4)];
        const v16bf bt =
            *(const v16bf*)&ktile[(ct << 10) + (ks << 9) + (lane << 4)];
        acc = __builtin_amdgcn_wmma_f32_16x16x32_bf16(
            false, a, false, bt, (short)0, acc, false, false);
      }
      const int sk = kb0 + (ct << 4) + lcol;
      const float mk = mask[b * SS + sk];
#pragma unroll
      for (int r = 0; r < 8; ++r) {
        const int lr = (rt << 4) + r + rhalf;
        const int sq = sq0 + lr;
        const float mq = mask[b * SS + sq];
        const float lg = acc[r] * 0.125f;  // 1/sqrt(64)
        qkOut[(bh * SS + sq) * SS + sk] = lg;
        const float mm = fminf(1.0f, mq + mk);
        logits[lr * SS + sk] = lg + mm * -1e9f;
      }
    }
    __syncthreads();
  }

  // ---- softmax over LDS logits: 4 threads per row, 256 cols each ----
  {
    const int row = tid >> 2;
    const int c0  = (tid & 3) << 8;
    float mx = -3.4e38f;
    for (int c = 0; c < 256; ++c) mx = fmaxf(mx, logits[row * SS + c0 + c]);
    red[tid] = mx;
    __syncthreads();
    mx = fmaxf(fmaxf(red[row * 4 + 0], red[row * 4 + 1]),
               fmaxf(red[row * 4 + 2], red[row * 4 + 3]));
    float sum = 0.f;
    for (int c = 0; c < 256; ++c) {
      const float e = __expf(logits[row * SS + c0 + c] - mx);
      logits[row * SS + c0 + c] = e;
      sum += e;
    }
    red[256 + tid] = sum;
    __syncthreads();
    sum = red[256 + row * 4 + 0] + red[256 + row * 4 + 1] +
          red[256 + row * 4 + 2] + red[256 + row * 4 + 3];
    const float inv = 1.0f / sum;
    const size_t rowOff = (bh * SS + sq0 + row) * SS;
    for (int c = 0; c < 256; ++c) {
      const float a = logits[row * SS + c0 + c] * inv;
      logits[row * SS + c0 + c] = a;
      attnOut[rowOff + c0 + c] = a;
    }
  }
  __syncthreads();

  // ---- z = attn @ V: attn A-frags from f32 LDS (2x v8f + cvt),
  //      V B-frags contiguous from packed tile ----
  v8f zacc[2] = {};
  for (int kk0 = 0; kk0 < SS; kk0 += 32) {
    stage_b128(Vp + ((kk0 >> 5) << 11) + (tid << 3), &vtile[tid << 3]);
    stage_wait();
    const float* rowp = &logits[((rt << 4) + lcol) * SS + kk0];
    const v8f lo = *(const v8f*)(rowp + rhalf);
    const v8f hi = *(const v8f*)(rowp + rhalf + 16);
    v16bf a;
#pragma unroll
    for (int j = 0; j < 8; ++j) {
      a[j] = (bf16)lo[j];
      a[j + 8] = (bf16)hi[j];
    }
#pragma unroll
    for (int c = 0; c < 2; ++c) {
      const v16bf bfr = *(const v16bf*)&vtile[((ctb + c) << 9) + (lane << 4)];
      zacc[c] = __builtin_amdgcn_wmma_f32_16x16x32_bf16(
          false, a, false, bfr, (short)0, zacc[c], false, false);
    }
    __syncthreads();
  }
#pragma unroll
  for (int c = 0; c < 2; ++c) {
    const int dep = ((ctb + c) << 4) + lcol;
#pragma unroll
    for (int r = 0; r < 8; ++r) {
      const int sq = sq0 + (rt << 4) + r + rhalf;
      zbf[(size_t)(b * SS + sq) * DMODEL + h * DEPTH + dep] = (bf16)zacc[c][r];
    }
  }
}

// ---------------------------------------------------------------------------
extern "C" void kernel_launch(void* const* d_in, const int* in_sizes, int n_in,
                              void* d_out, int out_size, void* d_ws,
                              size_t ws_size, hipStream_t stream) {
  const float* x    = (const float*)d_in[0];
  const float* mask = (const float*)d_in[1];
  const float* wq_w = (const float*)d_in[2];
  const float* wq_b = (const float*)d_in[3];
  const float* wk_w = (const float*)d_in[4];
  const float* wk_b = (const float*)d_in[5];
  const float* wv_w = (const float*)d_in[6];
  const float* wv_b = (const float*)d_in[7];
  const float* wo_w = (const float*)d_in[8];
  const float* wo_b = (const float*)d_in[9];

  // Outputs concatenated: out, attn, qk_values, q, k, v
  float* out     = (float*)d_out;
  float* attnOut = out + 8388608;        // B*S*D
  float* qkOut   = attnOut + 134217728;  // B*H*S*S
  float* qOut    = qkOut + 134217728;
  float* kOut    = qOut + 8388608;
  float* vOut    = kOut + 8388608;

  // Workspace (bf16 mirrors), ~88 MB total
  char* ws = (char*)d_ws;
  bf16* x_bf  = (bf16*)(ws);
  bf16* wq_pk = (bf16*)(ws + 16777216);
  bf16* wk_pk = (bf16*)(ws + 18874368);
  bf16* wv_pk = (bf16*)(ws + 20971520);
  bf16* wo_pk = (bf16*)(ws + 23068672);
  bf16* q_bf  = (bf16*)(ws + 25165824);
  bf16* k_bf  = (bf16*)(ws + 41943040);
  bf16* v_pk  = (bf16*)(ws + 58720256);
  bf16* z_bf  = (bf16*)(ws + 75497472);

  cast_f32_to_bf16<<<2048, 256, 0, stream>>>(x, x_bf, BB * SS * DMODEL);
  pack_w_bf16<<<256, 256, 0, stream>>>(wq_w, wq_pk);
  pack_w_bf16<<<256, 256, 0, stream>>>(wk_w, wk_pk);
  pack_w_bf16<<<256, 256, 0, stream>>>(wv_w, wv_pk);
  pack_w_bf16<<<256, 256, 0, stream>>>(wo_w, wo_pk);

  dim3 gg(64, 8, 1);  // M/128 x N/128
  gemm_bf16<<<gg, 256, 0, stream>>>(x_bf, wq_pk, wq_b, qOut, q_bf, 1, 1);
  gemm_bf16<<<gg, 256, 0, stream>>>(x_bf, wk_pk, wk_b, kOut, k_bf, 1, 1);
  gemm_bf16<<<gg, 256, 0, stream>>>(x_bf, wv_pk, wv_b, vOut, v_pk, 1, 2);

  dim3 ga(SS / 64, HH, BB);
  const size_t ldsBytes = 262144 + 8192 + 8192 + 4096 + 2048;  // 284672
  attn_kernel<<<ga, 256, ldsBytes, stream>>>(q_bf, k_bf, v_pk, mask, attnOut,
                                             qkOut, z_bf);

  gemm_bf16<<<gg, 256, 0, stream>>>(z_bf, wo_pk, wo_b, out, nullptr, 0, 0);
}
